// Attention_71631464563089
// MI455X (gfx1250) — matrix operand order repaired and checked by reading, not compiled, not run on hardware
//
#include <hip/hip_runtime.h>

// ---------------------------------------------------------------------------
// Types for gfx1250 WMMA (wave32).
// ---------------------------------------------------------------------------
typedef __attribute__((ext_vector_type(16))) __bf16 v16bf;
typedef __attribute__((ext_vector_type(8)))  __bf16 bf8;
typedef __attribute__((ext_vector_type(8)))  float  v8f;

#define H_    16
#define DH_   64
#define D_    1024
#define QB_   8
#define N_    1024
#define MTOT  (QB_ * N_)   // 8192 rows

#if defined(__has_builtin)
#if __has_builtin(__builtin_amdgcn_tensor_load_to_lds)
#define HAVE_TDM 1
#endif
#endif
#ifndef HAVE_TDM
#define HAVE_TDM 0
#endif

__device__ __forceinline__ v8f vzero8() {
  v8f z;
#pragma unroll
  for (int i = 0; i < 8; ++i) z[i] = 0.0f;
  return z;
}

// Low 32 bits of a generic pointer into the LDS aperture == LDS byte offset.
__device__ __forceinline__ unsigned lds_off(const void* p) {
  return (unsigned)(unsigned long long)p;
}

#if HAVE_TDM
// Tensor Data Mover: DMA a 2D tile (rows x cols elements, bf16) from global
// (row stride = stride elements) into LDS, packed contiguously.
// D# packing per CDNA5 ISA 8.3/8.4 (group0: count/lds/global/type=2;
// group1: data_size=2B, tensor dims == tile dims so no OOB, dim0 stride).
__device__ __forceinline__ void tdm_load_2d(unsigned lds_addr, const void* gaddr,
                                            unsigned rows, unsigned cols,
                                            unsigned stride) {
  typedef unsigned int u32x4 __attribute__((ext_vector_type(4)));
  typedef int i32x4 __attribute__((ext_vector_type(4)));
  typedef int i32x8 __attribute__((ext_vector_type(8)));
  unsigned long long ga = (unsigned long long)gaddr;
  u32x4 g0;
  g0[0] = 1u;                                    // count=1, user descriptor
  g0[1] = lds_addr;                              // LDS byte address
  g0[2] = (unsigned)(ga & 0xffffffffu);          // global_addr[31:0]
  g0[3] = (unsigned)((ga >> 32) & 0x01ffffffu)   // global_addr[56:32]
          | (2u << 30);                          // type=2 ("image")
  i32x8 g1;
  g1[0] = 0x00010000;                            // data_size=1 (2 bytes)
  g1[1] = (int)((cols & 0xffffu) << 16);         // tensor_dim0[15:0]
  g1[2] = (int)(((cols >> 16) & 0xffffu) | ((rows & 0xffffu) << 16));   // td0 hi | td1 lo
  g1[3] = (int)(((rows >> 16) & 0xffffu) | ((cols & 0xffffu) << 16));   // td1 hi | tile_dim0
  g1[4] = (int)(rows & 0xffffu);                 // tile_dim1 (tile_dim2=0)
  g1[5] = (int)stride;                           // tensor_dim0_stride[31:0]
  g1[6] = 0;
  g1[7] = 0;
  i32x4 z4 = {0, 0, 0, 0};
#if __clang_major__ >= 23
  i32x8 z8 = {0, 0, 0, 0, 0, 0, 0, 0};
  __builtin_amdgcn_tensor_load_to_lds(g0, g1, z4, z4, z8, 0);
#else
  __builtin_amdgcn_tensor_load_to_lds(g0, g1, z4, z4, 0);
#endif
}
#endif  // HAVE_TDM

// Async copy of 16 bytes global -> LDS (per-lane), tracked by ASYNCcnt.
__device__ __forceinline__ void async_b128(unsigned lds_addr, const void* gaddr) {
  asm volatile("global_load_async_to_lds_b128 %0, %1, off"
               :: "v"(lds_addr), "v"(gaddr) : "memory");
}
__device__ __forceinline__ void wait_asynccnt0() {
  asm volatile("s_wait_asynccnt 0" ::: "memory");
}

// ---------------------------------------------------------------------------
// Cross-lane reductions via ds_swizzle_b32 xor mode (ISA 11.1/11.2):
// offset = xor_mask<<10 | or_mask<<5 | and_mask; and_mask=0x1f keeps all lanes.
// One DS op per step -- no bounds-check VALU like __shfl_xor emits.
// ---------------------------------------------------------------------------
template <int XMASK>
__device__ __forceinline__ float swz_xor(float v) {
  return __int_as_float(
      __builtin_amdgcn_ds_swizzle(__float_as_int(v), (XMASK << 10) | 0x1f));
}
__device__ __forceinline__ float half_max(float v) {   // max over 16-lane halves
  v = fmaxf(v, swz_xor<1>(v));
  v = fmaxf(v, swz_xor<2>(v));
  v = fmaxf(v, swz_xor<4>(v));
  v = fmaxf(v, swz_xor<8>(v));
  return v;
}
__device__ __forceinline__ float half_sum(float v) {   // sum over 16-lane halves
  v += swz_xor<1>(v);
  v += swz_xor<2>(v);
  v += swz_xor<4>(v);
  v += swz_xor<8>(v);
  return v;
}
__device__ __forceinline__ float full_sum(float v) {   // sum over 32 lanes
  v += swz_xor<1>(v);
  v += swz_xor<2>(v);
  v += swz_xor<4>(v);
  v += swz_xor<8>(v);
  v += swz_xor<16>(v);
  return v;
}

// A-fragment (16x32 bf16, MxK): slots 0..7 = K=kb..kb+7, slots 8..15 = K=kb+16..kb+23
// (kb = 0 for lanes 0-15, 8 for lanes 16-31); caller passes the two 16B chunks.
__device__ __forceinline__ v16bf ld_frag(const __bf16* lo, const __bf16* hi) {
  bf8 a = *(const bf8*)lo;
  bf8 b = *(const bf8*)hi;
  v16bf r;
#pragma unroll
  for (int i = 0; i < 8; ++i) { r[i] = a[i]; r[i + 8] = b[i]; }
  return r;
}
// B-fragment (32x16 bf16, KxN): slots 0..15 = 16 consecutive K values for this lane's column.
__device__ __forceinline__ v16bf ld_frag16(const __bf16* p) { return ld_frag(p, p + 8); }

// ---------------------------------------------------------------------------
// Kernel: fp32 -> bf16 convert (for w_in / w_out).
// ---------------------------------------------------------------------------
__global__ void cvt_bf16_kernel(const float* __restrict__ in, __bf16* __restrict__ out, int n) {
  int i = blockIdx.x * 256 + threadIdx.x;
  if (i < n) out[i] = (__bf16)in[i];
}

// ---------------------------------------------------------------------------
// Kernel: row LayerNorm over D=1024, fp32 in -> bf16 out. One block per row.
// ---------------------------------------------------------------------------
__global__ void ln_rows_kernel(const float* __restrict__ x, const float* __restrict__ g,
                               const float* __restrict__ b, __bf16* __restrict__ out) {
  int row = blockIdx.x;
  int tid = threadIdx.x;
  const float* xr = x + (size_t)row * D_;
  float vals[4];
  float s = 0.f, s2 = 0.f;
#pragma unroll
  for (int i = 0; i < 4; ++i) {
    float t = xr[tid + i * 256];
    vals[i] = t; s += t; s2 += t * t;
  }
  s = full_sum(s); s2 = full_sum(s2);
  __shared__ float rs[8], rs2[8];
  if ((tid & 31) == 0) { rs[tid >> 5] = s; rs2[tid >> 5] = s2; }
  __syncthreads();
  s = 0.f; s2 = 0.f;
#pragma unroll
  for (int i = 0; i < 8; ++i) { s += rs[i]; s2 += rs2[i]; }
  float mean = s * (1.0f / D_);
  float var  = s2 * (1.0f / D_) - mean * mean;
  float rstd = rsqrtf(var + 1e-5f);
#pragma unroll
  for (int i = 0; i < 4; ++i) {
    int c = tid + i * 256;
    out[(size_t)row * D_ + c] = (__bf16)((vals[i] - mean) * rstd * g[c] + b[c]);
  }
}

// ---------------------------------------------------------------------------
// Generic bf16 WMMA GEMM: C[M,Nout] = A[M,K] @ Brow[Nout,K]^T.
// 256 threads = 8 waves; block tile 128x128; wave tile 32x64; BK=64.
// Tiles (128x64 bf16 = 16KB) are DMA'd into LDS by the Tensor Data Mover,
// double-buffered on TENSORcnt: WMMA chews the current buffer (16 wmma ops)
// while the TDM fills the next one.
// MODE 0: fp32 store + bias.     (final output projection)
// MODE 1: bf16 store, layout [h][qb][n][dh]   (f_q / f_k)
// MODE 2: bf16 store, layout [h][qb][dh][n]   (f_v transposed)
// ---------------------------------------------------------------------------
template <int MODE>
__global__ void gemm_bf16_kernel(const __bf16* __restrict__ A, const __bf16* __restrict__ B,
                                 float* __restrict__ outF, __bf16* __restrict__ outB,
                                 const float* __restrict__ bias, int M, int Nout, int K) {
  __shared__ __align__(16) __bf16 As[2][128][64];
  __shared__ __align__(16) __bf16 Bs[2][128][64];
  int tid = threadIdx.x, lane = tid & 31, wid = tid >> 5;
  int wm = wid & 3, wn = wid >> 2;
  int l15 = lane & 15;
  int lh  = (lane < 16) ? 0 : 1;

  v8f acc[2][4];
#pragma unroll
  for (int mi = 0; mi < 2; ++mi)
#pragma unroll
    for (int ni = 0; ni < 4; ++ni) acc[mi][ni] = vzero8();

  size_t abase = (size_t)(blockIdx.y * 128) * K;
  size_t bbase = (size_t)(blockIdx.x * 128) * K;

#if HAVE_TDM
  if (tid == 0) {   // TDM ignores EXEC: one DMA pair issued from wave 0
    tdm_load_2d(lds_off(&As[0][0][0]), A + abase, 128, 64, (unsigned)K);
    tdm_load_2d(lds_off(&Bs[0][0][0]), B + bbase, 128, 64, (unsigned)K);
  }
#else
  int rA = tid >> 1, cA = (tid & 1) * 32;
#endif

  int buf = 0;
  for (int k0 = 0; k0 < K; k0 += 64) {
#if HAVE_TDM
    if (wid == 0) __builtin_amdgcn_s_wait_tensorcnt(0);
    __syncthreads();  // current buffer ready for everyone
    if (tid == 0 && k0 + 64 < K) {
      tdm_load_2d(lds_off(&As[buf ^ 1][0][0]), A + abase + k0 + 64, 128, 64, (unsigned)K);
      tdm_load_2d(lds_off(&Bs[buf ^ 1][0][0]), B + bbase + k0 + 64, 128, 64, (unsigned)K);
    }
#else
    const __bf16* ga = A + abase + (size_t)rA * K + k0 + cA;
    const __bf16* gb = B + bbase + (size_t)rA * K + k0 + cA;
#pragma unroll
    for (int c = 0; c < 4; ++c) {
      *(bf8*)&As[buf][rA][cA + 8 * c] = *(const bf8*)(ga + 8 * c);
      *(bf8*)&Bs[buf][rA][cA + 8 * c] = *(const bf8*)(gb + 8 * c);
    }
    __syncthreads();
    if (k0 + 64 < K) {
      __builtin_prefetch(ga + 64, 0, 0);
      __builtin_prefetch(gb + 64, 0, 0);
    }
#endif
#pragma unroll
    for (int kk = 0; kk < 64; kk += 32) {
      v16bf af[2];
#pragma unroll
      for (int mi = 0; mi < 2; ++mi) {
        int rr = wm * 32 + mi * 16 + l15;
        int kbA = kk + lh * 8;
        af[mi] = ld_frag(&As[buf][rr][kbA], &As[buf][rr][kbA + 16]);
      }
#pragma unroll
      for (int ni = 0; ni < 4; ++ni) {
        int cc  = wn * 64 + ni * 16 + l15;
        int kb2 = kk + lh * 8;
        v16bf bfrag = ld_frag(&Bs[buf][cc][kb2], &Bs[buf][cc][kb2 + 16]);
#pragma unroll
        for (int mi = 0; mi < 2; ++mi)
          acc[mi][ni] = __builtin_amdgcn_wmma_f32_16x16x32_bf16(
              false, af[mi], false, bfrag, (short)0, acc[mi][ni], false, false);
      }
    }
    __syncthreads();
    buf ^= 1;
  }

#pragma unroll
  for (int mi = 0; mi < 2; ++mi)
#pragma unroll
    for (int ni = 0; ni < 4; ++ni) {
      int col = blockIdx.x * 128 + wn * 64 + ni * 16 + l15;
#pragma unroll
      for (int r = 0; r < 8; ++r) {
        int m = blockIdx.y * 128 + wm * 32 + mi * 16 + r + lh * 8;
        float vv = acc[mi][ni][r];
        if (MODE == 0) {
          outF[(size_t)m * Nout + col] = vv + bias[col];
        } else {
          int hh = col >> 6, dh = col & 63, qq = m >> 10, nn = m & 1023;
          size_t idx;
          if (MODE == 1) idx = (((size_t)(hh * 8 + qq) * N_ + nn) * DH_ + dh);
          else           idx = (((size_t)(hh * 8 + qq) * DH_ + dh) * N_ + nn);
          outB[idx] = (__bf16)vv;
        }
      }
    }
}

// ---------------------------------------------------------------------------
// inv(): y = f @ w_inv^T + b_inv, then LN over DH; plus var/cov stats and
// weight_factor = sigmoid(dyn*(vw*(var_q+var_k)+cw*cov)).
// One wave per row (32 lanes x 2 columns); 8 waves/block, 8 rows/wave.
// ---------------------------------------------------------------------------
__global__ void inv_stats_kernel(const __bf16* __restrict__ fq, const __bf16* __restrict__ fk,
                                 const float* __restrict__ w_inv, const float* __restrict__ b_inv,
                                 const float* __restrict__ g, const float* __restrict__ bl,
                                 const float* __restrict__ dyn_w, const float* __restrict__ var_w,
                                 const float* __restrict__ cov_w,
                                 __bf16* __restrict__ fq_inv, __bf16* __restrict__ fk_inv,
                                 float* __restrict__ wf) {
  __shared__ float W[64][65];
  __shared__ float Bv[64], G[64], BL[64];
  __shared__ float aq[8][64], ak[8][64];
  int tid = threadIdx.x, lane = tid & 31, wid = tid >> 5;
  for (int i = tid; i < 4096; i += 256) W[i >> 6][i & 63] = w_inv[i];
  if (tid < 64) { Bv[tid] = b_inv[tid]; G[tid] = g[tid]; BL[tid] = bl[tid]; }
  __syncthreads();
  float dw = dyn_w[0], vw = var_w[0], cw = cov_w[0];
  int j0 = lane * 2, j1 = lane * 2 + 1;

  for (int it = 0; it < 8; ++it) {
    size_t row = (size_t)blockIdx.x * 64 + wid * 8 + it;  // [H][QB][N] flat
    const __bf16* qr = fq + row * DH_;
    const __bf16* kr = fk + row * DH_;
    float q0 = (float)qr[j0], q1 = (float)qr[j1];
    float k0 = (float)kr[j0], k1 = (float)kr[j1];
    aq[wid][j0] = q0; aq[wid][j1] = q1;
    ak[wid][j0] = k0; ak[wid][j1] = k1;
    float sq  = full_sum(q0 + q1);
    float sqq = full_sum(q0 * q0 + q1 * q1);
    float sk  = full_sum(k0 + k1);
    float skk = full_sum(k0 * k0 + k1 * k1);
    float sqk = full_sum(q0 * k0 + q1 * k1);
    const float inv64 = 1.0f / 64.0f;
    float qm = sq * inv64, km = sk * inv64;
    float varq = sqq * inv64 - qm * qm;
    float vark = skk * inv64 - km * km;
    float cov  = sqk * inv64 - qm * km;
    float z = dw * (vw * (varq + vark) + cw * cov);
    if (lane == 0) wf[row] = 1.0f / (1.0f + __expf(-z));

    // --- q branch: y = aq @ W^T + b; LN; store ---
    {
      float y0 = Bv[j0], y1 = Bv[j1];
#pragma unroll
      for (int t = 0; t < 64; ++t) {
        float a = aq[wid][t];
        y0 += a * W[j0][t];
        y1 += a * W[j1][t];
      }
      float sy  = full_sum(y0 + y1) * inv64;
      float syy = full_sum(y0 * y0 + y1 * y1) * inv64;
      float rstd = rsqrtf(syy - sy * sy + 1e-5f);
      fq_inv[row * DH_ + j0] = (__bf16)((y0 - sy) * rstd * G[j0] + BL[j0]);
      fq_inv[row * DH_ + j1] = (__bf16)((y1 - sy) * rstd * G[j1] + BL[j1]);
    }
    // --- k branch ---
    {
      float y0 = Bv[j0], y1 = Bv[j1];
#pragma unroll
      for (int t = 0; t < 64; ++t) {
        float a = ak[wid][t];
        y0 += a * W[j0][t];
        y1 += a * W[j1][t];
      }
      float sy  = full_sum(y0 + y1) * inv64;
      float syy = full_sum(y0 * y0 + y1 * y1) * inv64;
      float rstd = rsqrtf(syy - sy * sy + 1e-5f);
      fk_inv[row * DH_ + j0] = (__bf16)((y0 - sy) * rstd * G[j0] + BL[j0]);
      fk_inv[row * DH_ + j1] = (__bf16)((y1 - sy) * rstd * G[j1] + BL[j1]);
    }
  }
}

// ---------------------------------------------------------------------------
// Flash attention per (h,qb). Block = 128 query rows (8 waves x 16 rows),
// KV tiles of 32, double-buffered via GLOBAL_LOAD_ASYNC_TO_LDS (ASYNCcnt).
// WMMA for QK^T and P*V; online softmax with ds_swizzle half-wave reductions.
// out written as [qb][n][h*64+dh] bf16 (feeds output projection GEMM).
// ---------------------------------------------------------------------------
__global__ void flash_kernel(const __bf16* __restrict__ fqi, const __bf16* __restrict__ fki,
                             const __bf16* __restrict__ fvt, const float* __restrict__ wf,
                             __bf16* __restrict__ aout) {
  __shared__ __align__(16) __bf16 Kt[2][32][64];   // K tile  [kv][dh]
  __shared__ __align__(16) __bf16 Vt[2][64][32];   // V tile  [dh][kv]
  __shared__ __align__(16) __bf16 Pst[8][16][32];  // per-wave P staging
  int tid = threadIdx.x, lane = tid & 31, wid = tid >> 5;
  int hq = blockIdx.y, h = hq >> 3, qb = hq & 7;
  const __bf16* FQ = fqi + (size_t)hq * N_ * DH_;
  const __bf16* FK = fki + (size_t)hq * N_ * DH_;
  const __bf16* FV = fvt + (size_t)hq * DH_ * N_;
  const float*  WF = wf + (size_t)hq * N_;
  int r0  = blockIdx.x * 128 + wid * 16;
  int l15 = lane & 15;
  int lh  = (lane < 16) ? 0 : 1;
  int kbA = lh * 8;

  // Cooperative async tile fill: Kt 32x64 and Vt 64x32 (16B per thread each).
  int rK = tid >> 3, cK = (tid & 7) * 8;
  int rV = tid >> 2, cV = (tid & 3) * 8;
  auto issue_tiles = [&](int b, int kb) {
    async_b128(lds_off(&Kt[b][rK][cK]), FK + (size_t)(kb + rK) * DH_ + cK);
    async_b128(lds_off(&Vt[b][rV][cV]), FV + (size_t)rV * N_ + kb + cV);
  };
  issue_tiles(0, 0);

  // Q fragments stay in registers for the whole kernel.
  const __bf16* qp = FQ + (size_t)(r0 + l15) * DH_;
  v16bf qa0 = ld_frag(qp + kbA, qp + kbA + 16);            // K-dim 0..31
  v16bf qa1 = ld_frag(qp + 32 + kbA, qp + 32 + kbA + 16);  // K-dim 32..63

  float wfv[8], mrun[8], lrun[8];
  v8f o[4];
#pragma unroll
  for (int d = 0; d < 4; ++d) o[d] = vzero8();
#pragma unroll
  for (int r = 0; r < 8; ++r) {
    wfv[r]  = WF[r0 + r + lh * 8] * 0.125f;  // scale = DH^-0.5 folded in
    mrun[r] = -1e30f;
    lrun[r] = 0.0f;
  }

  int buf = 0;
  for (int kb = 0; kb < N_; kb += 32) {
    wait_asynccnt0();   // this wave's tile stores have landed in LDS
    __syncthreads();    // ...and everyone else's too
    if (kb + 32 < N_) issue_tiles(buf ^ 1, kb + 32);

    // S = Q (16x64) * K^T (64x32): two 16-col sub-tiles
    v8f s[2];
#pragma unroll
    for (int ns = 0; ns < 2; ++ns) {
      int kc  = ns * 16 + l15;  // kv column within tile
      int kb2 = lh * 8;
      v16bf b0 = ld_frag(&Kt[buf][kc][kb2], &Kt[buf][kc][kb2 + 16]);       // dh 0..31
      v16bf b1 = ld_frag(&Kt[buf][kc][32 + kb2], &Kt[buf][kc][48 + kb2]);  // dh 32..63
      v8f a = vzero8();
      a = __builtin_amdgcn_wmma_f32_16x16x32_bf16(false, qa0, false, b0, (short)0, a, false, false);
      a = __builtin_amdgcn_wmma_f32_16x16x32_bf16(false, qa1, false, b1, (short)0, a, false, false);
      s[ns] = a;
    }

    // online softmax + stage P (bf16) into per-wave LDS
#pragma unroll
    for (int r = 0; r < 8; ++r) {
      float x0 = s[0][r] * wfv[r];
      float x1 = s[1][r] * wfv[r];
      float mt = half_max(fmaxf(x0, x1));
      float mn = fmaxf(mrun[r], mt);
      float al = __expf(mrun[r] - mn);
      mrun[r] = mn;
      float p0 = __expf(x0 - mn);
      float p1 = __expf(x1 - mn);
      lrun[r] = lrun[r] * al + half_sum(p0 + p1);
#pragma unroll
      for (int d = 0; d < 4; ++d) o[d][r] = o[d][r] * al;
      int pr = r + lh * 8;
      Pst[wid][pr][l15]      = (__bf16)p0;
      Pst[wid][pr][16 + l15] = (__bf16)p1;
    }

    // O += P (16x32) * V (32x64)
    v16bf pa = ld_frag(&Pst[wid][l15][kbA], &Pst[wid][l15][kbA + 16]);
#pragma unroll
    for (int d = 0; d < 4; ++d) {
      int vcol = d * 16 + l15;
      int kb2  = lh * 8;
      v16bf bv = ld_frag(&Vt[buf][vcol][kb2], &Vt[buf][vcol][kb2 + 16]);
      o[d] = __builtin_amdgcn_wmma_f32_16x16x32_bf16(false, pa, false, bv, (short)0, o[d], false, false);
    }
    __syncthreads();
    buf ^= 1;
  }

  // epilogue: divide by l, store [qb][n][h*64+dh]
#pragma unroll
  for (int r = 0; r < 8; ++r) {
    float il = 1.0f / lrun[r];
    int m = r0 + r + lh * 8;
    size_t rowbase = ((size_t)(qb * N_ + m)) * (H_ * DH_) + h * DH_;
#pragma unroll
    for (int d = 0; d < 4; ++d)
      aout[rowbase + d * 16 + l15] = (__bf16)(o[d][r] * il);
  }
}

// ---------------------------------------------------------------------------
// Launch
// ---------------------------------------------------------------------------
extern "C" void kernel_launch(void* const* d_in, const int* in_sizes, int n_in,
                              void* d_out, int out_size, void* d_ws, size_t ws_size,
                              hipStream_t stream) {
  const float* q        = (const float*)d_in[0];
  const float* k        = (const float*)d_in[1];
  const float* v        = (const float*)d_in[2];
  const float* ln_in_g  = (const float*)d_in[3];
  const float* ln_in_b  = (const float*)d_in[4];
  const float* w_in     = (const float*)d_in[5];
  const float* w_out    = (const float*)d_in[6];
  const float* b_out    = (const float*)d_in[7];
  const float* w_inv    = (const float*)d_in[8];
  const float* b_inv    = (const float*)d_in[9];
  const float* ln_inv_g = (const float*)d_in[10];
  const float* ln_inv_b = (const float*)d_in[11];
  const float* dyn_w    = (const float*)d_in[12];
  const float* var_w    = (const float*)d_in[13];
  const float* cov_w    = (const float*)d_in[14];

  char* ws = (char*)d_ws;
  const size_t MB = 1ull << 20;
  // Region reuse: [0,48MB) holds xn_q/k/v first, then fq_inv/fk_inv/attn.
  __bf16* xn_q   = (__bf16*)(ws + 0 * MB);
  __bf16* xn_k   = (__bf16*)(ws + 16 * MB);
  __bf16* xn_v   = (__bf16*)(ws + 32 * MB);
  __bf16* fq_inv = (__bf16*)(ws + 0 * MB);
  __bf16* fk_inv = (__bf16*)(ws + 16 * MB);
  __bf16* attn   = (__bf16*)(ws + 32 * MB);
  __bf16* winb   = (__bf16*)(ws + 48 * MB);
  __bf16* woutb  = (__bf16*)(ws + 50 * MB);
  float*  wfbuf  = (float*)(ws + 52 * MB);
  __bf16* f_q    = (__bf16*)(ws + 53 * MB);
  __bf16* f_k    = (__bf16*)(ws + 69 * MB);
  __bf16* f_vt   = (__bf16*)(ws + 85 * MB);

  // 1) weight conversion
  cvt_bf16_kernel<<<4096, 256, 0, stream>>>(w_in, winb, D_ * H_ * DH_);
  cvt_bf16_kernel<<<4096, 256, 0, stream>>>(w_out, woutb, D_ * H_ * DH_);

  // 2) input LayerNorms
  ln_rows_kernel<<<MTOT, 256, 0, stream>>>(q, ln_in_g, ln_in_b, xn_q);
  ln_rows_kernel<<<MTOT, 256, 0, stream>>>(k, ln_in_g, ln_in_b, xn_k);
  ln_rows_kernel<<<MTOT, 256, 0, stream>>>(v, ln_in_g, ln_in_b, xn_v);

  // 3) input projections (WMMA GEMMs, TDM-fed)
  dim3 gg(D_ / 128, MTOT / 128);
  gemm_bf16_kernel<1><<<gg, 256, 0, stream>>>(xn_q, winb, nullptr, f_q, nullptr, MTOT, D_, D_);
  gemm_bf16_kernel<1><<<gg, 256, 0, stream>>>(xn_k, winb, nullptr, f_k, nullptr, MTOT, D_, D_);
  gemm_bf16_kernel<2><<<gg, 256, 0, stream>>>(xn_v, winb, nullptr, f_vt, nullptr, MTOT, D_, D_);

  // 4) inv() + stats + weight factor
  inv_stats_kernel<<<2048, 256, 0, stream>>>(f_q, f_k, w_inv, b_inv, ln_inv_g, ln_inv_b,
                                             dyn_w, var_w, cov_w, fq_inv, fk_inv, wfbuf);

  // 5) flash attention (async-LDS fed)
  flash_kernel<<<dim3(N_ / 128, H_ * QB_), 256, 0, stream>>>(fq_inv, fk_inv, f_vt, wfbuf, attn);

  // 6) output projection -> fp32 d_out
  gemm_bf16_kernel<0><<<gg, 256, 0, stream>>>(attn, woutb, (float*)d_out, nullptr, b_out, MTOT, D_, D_);
}
